// DQNv2_28784870818381
// MI455X (gfx1250) — compile-verified, matrix operand order checked by reading, not run on hardware
//
#include <hip/hip_runtime.h>
#include <stdint.h>

#define NNODES 50000
#define NEDGES 800000
#define NB 32
#define NA 10

typedef __attribute__((ext_vector_type(16))) __bf16 v16bf;
typedef __attribute__((ext_vector_type(8)))  __bf16 v8bf;
typedef __attribute__((ext_vector_type(8)))  float  v8f;
typedef unsigned int u32x4 __attribute__((ext_vector_type(4)));
typedef int          i32x4 __attribute__((ext_vector_type(4)));
typedef int          i32x8 __attribute__((ext_vector_type(8)));

// ---------- helpers ----------
__device__ __forceinline__ unsigned short f2bf(float f) {
  unsigned u = __float_as_uint(f);
  u += 0x7FFFu + ((u >> 16) & 1u);            // round to nearest even
  return (unsigned short)(u >> 16);
}
// order-preserving float <-> uint map (for atomicMax-based segment max)
__device__ __forceinline__ unsigned ford(float f) {
  unsigned u = __float_as_uint(f);
  return (u & 0x80000000u) ? ~u : (u | 0x80000000u);
}
__device__ __forceinline__ float funord(unsigned u) {
  return __uint_as_float((u & 0x80000000u) ? (u & 0x7FFFFFFFu) : ~u);
}
__device__ __forceinline__ float wave_sum(float v) {
#pragma unroll
  for (int off = 16; off; off >>= 1) v += __shfl_xor(v, off, 32);
  return v;
}

// ---------- generic fill ----------
__global__ void fill_u32_kernel(unsigned* p, unsigned v, long long n) {
  long long i = (long long)blockIdx.x * blockDim.x + threadIdx.x;
  long long s = (long long)gridDim.x * blockDim.x;
  for (; i < n; i += s) p[i] = v;
}

// ---------- f32 -> bf16 (row-major, K zero-padded to Kp) ----------
__global__ void cvt_pad_bf16_kernel(const float* __restrict__ src,
                                    unsigned short* __restrict__ dst,
                                    int M, int K, int Kp) {
  long long n = (long long)M * Kp;
  long long i = (long long)blockIdx.x * blockDim.x + threadIdx.x;
  long long s = (long long)gridDim.x * blockDim.x;
  for (; i < n; i += s) {
    int m = (int)(i / Kp), k = (int)(i % Kp);
    dst[i] = (k < K) ? f2bf(src[(long long)m * K + k]) : (unsigned short)0;
  }
}

// ---------- W[K x N] (row-major) -> Wt[Np x Kp] bf16, zero padded ----------
__global__ void cvt_transpose_bf16_kernel(const float* __restrict__ W,
                                          unsigned short* __restrict__ Wt,
                                          int K, int N, int Np, int Kp) {
  long long total = (long long)Np * Kp;
  long long i = (long long)blockIdx.x * blockDim.x + threadIdx.x;
  long long s = (long long)gridDim.x * blockDim.x;
  for (; i < total; i += s) {
    int n = (int)(i / Kp), k = (int)(i % Kp);
    Wt[i] = (n < N && k < K) ? f2bf(W[(long long)k * N + n]) : (unsigned short)0;
  }
}

// ---------- WMMA GEMM with LDS-staged weight panel ----------
// C[M x Nout] = A[M x Kp] * Bt[Nout x Kp]^T  (+bias, optional relu, optional bf16 copy)
// blockDim = 256 (8 waves). grid.x = Nout/(NT*16) n-strips; grid.y = m chunks.
// The NT*16-column weight panel (contiguous rows of Bt) is pulled into LDS:
// a TDM tensor_load_to_lds is issued as an async LDS prefetch, then a
// vectorized b128 cooperative copy writes the authoritative data. Each wave
// then grid-strides over 16-row m-tiles: one A fragment feeds NT WMMAs.
template <int NT>
__global__ __launch_bounds__(256) void gemm_bf16_wmma_kernel(
    const unsigned short* __restrict__ A,
    const unsigned short* __restrict__ Bt,
    const float* __restrict__ bias,
    float* __restrict__ C,
    unsigned short* __restrict__ Cbf,
    int M, int Kp, int Nvalid, int ldC, int ldCbf, int relu) {
  __shared__ __align__(16) unsigned short smem[64 * 256];  // up to 64 cols x Kp=256
  const int tid  = threadIdx.x;
  const int lane = tid & 31;
  const int wave = tid >> 5;
  const int n0   = blockIdx.x * (NT * 16);
  const int panel = NT * 16 * Kp;                 // ushorts, contiguous in Bt
  const unsigned short* Bpanel = Bt + (size_t)n0 * Kp;

  // --- TDM async tensor prefetch of the panel into LDS (1 x panel/2 dwords) ---
  if (wave == 0) {
    unsigned lds_off = (unsigned)(size_t)(&smem[0]);      // low 32 bits = LDS offset
    unsigned long long ga = (unsigned long long)(size_t)Bpanel;
    unsigned dwords = (unsigned)(panel >> 1);             // 4-byte elements
    u32x4 g0;
    g0[0] = 1u;                                           // count=1, user mode
    g0[1] = lds_off;                                      // lds_addr
    g0[2] = (unsigned)(ga & 0xFFFFFFFFu);                 // global_addr[31:0]
    g0[3] = (unsigned)((ga >> 32) & 0x01FFFFFFu) | (2u << 30);  // addr[56:32] | type=2
    i32x8 g1;
    g1[0] = (int)(2u << 16);                              // data_size=4B, no flags
    g1[1] = (int)((dwords & 0xFFFFu) << 16);              // tensor_dim0[15:0]
    g1[2] = (int)(((dwords >> 16) & 0xFFFFu) | (1u << 16)); // tensor_dim0[31:16] | tensor_dim1=1
    g1[3] = (int)((dwords & 0xFFFFu) << 16);              // tile_dim0 = dwords (<=8192)
    g1[4] = 1;                                            // tile_dim1 = 1
    g1[5] = (int)dwords;                                  // tensor_dim0_stride[31:0]
    g1[6] = 0; g1[7] = 0;
    i32x4 gz4; gz4[0] = 0; gz4[1] = 0; gz4[2] = 0; gz4[3] = 0;
    i32x8 gz8;
#pragma unroll
    for (int i = 0; i < 8; ++i) gz8[i] = 0;
    __builtin_amdgcn_tensor_load_to_lds(g0, g1, gz4, gz4, gz8, 0);
    __builtin_amdgcn_s_wait_tensorcnt(0);
  }
  __syncthreads();

  // --- authoritative vectorized staging (b128 per lane) ---
  {
    const uint4* gsrc = (const uint4*)(const void*)Bpanel;
    uint4* ldst = (uint4*)(void*)smem;
    const int chunks = panel >> 3;                        // 8 ushorts per uint4
    for (int i = tid; i < chunks; i += 256) ldst[i] = gsrc[i];
  }
  __syncthreads();

  const int half = lane >> 4;   // 0: K kh..kh+7 & 16+kh.. ; 1: shifted by 8
  const int l15  = lane & 15;
  const int mtiles = (M + 15) >> 4;
  const int mstride = gridDim.y * 8;

  for (int mt = blockIdx.y * 8 + wave; mt < mtiles; mt += mstride) {
    const int m0 = mt * 16;
    int mrow = m0 + l15;
    if (mrow >= M) mrow = M - 1;                       // clamp (store guarded)
    const unsigned short* Ap = A + (size_t)mrow * Kp + half * 8;

    // prefetch next m-tile's A row into cache while computing this one
    const int mt2 = mt + mstride;
    if (mt2 < mtiles)
      __builtin_prefetch(A + ((size_t)(mt2 * 16 + l15)) * Kp, 0, 1);

    v8f acc[NT];
#pragma unroll
    for (int j = 0; j < NT; ++j)
#pragma unroll
      for (int r = 0; r < 8; ++r) acc[j][r] = 0.0f;

    for (int k = 0; k < Kp; k += 32) {
      v8bf a0 = *(const v8bf*)(const void*)(Ap + k);
      v8bf a1 = *(const v8bf*)(const void*)(Ap + k + 16);
      // batch all B-fragment LDS loads before the WMMA quartet
      v8bf bl[NT], bh[NT];
#pragma unroll
      for (int j = 0; j < NT; ++j) {
        const unsigned short* Bp = &smem[(16 * j + l15) * Kp + half * 8 + k];
        bl[j] = *(const v8bf*)(const void*)(Bp);
        bh[j] = *(const v8bf*)(const void*)(Bp + 16);
      }
      v16bf av;
#pragma unroll
      for (int i = 0; i < 8; ++i) { av[i] = a0[i]; av[i + 8] = a1[i]; }
#pragma unroll
      for (int j = 0; j < NT; ++j) {
        v16bf bv;
#pragma unroll
        for (int i = 0; i < 8; ++i) { bv[i] = bl[j][i]; bv[i + 8] = bh[j][i]; }
        acc[j] = __builtin_amdgcn_wmma_f32_16x16x32_bf16(
            false, av, false, bv, (short)0, acc[j], false, false);
      }
    }

    const int rbase = m0 + half * 8;
#pragma unroll
    for (int j = 0; j < NT; ++j) {
      const int col = n0 + 16 * j + l15;
      if (col < Nvalid) {
        const float bb = bias ? bias[col] : 0.0f;
#pragma unroll
        for (int r = 0; r < 8; ++r) {
          const int row = rbase + r;
          if (row < M) {
            float v = acc[j][r] + bb;
            if (relu && v < 0.0f) v = 0.0f;
            C[(size_t)row * ldC + col] = v;
            if (Cbf) Cbf[(size_t)row * ldCbf + col] = f2bf(v);
          }
        }
      }
    }
  }
}

// ---------- edge phase ----------
// one wave32 per (edge, head): coalesced channel reads + shuffle reduction
__global__ void edge_logit_kernel(const float* __restrict__ xl, const float* __restrict__ xr,
                                  const float* __restrict__ ea, const float* __restrict__ We,
                                  const float* __restrict__ att,
                                  const int* __restrict__ src, const int* __restrict__ dst,
                                  float* __restrict__ elog, unsigned* __restrict__ emaxu,
                                  int E, int H, int C) {
  int w = (int)((blockIdx.x * (long long)blockDim.x + threadIdx.x) >> 5);
  int lane = threadIdx.x & 31;
  if (w >= E * H) return;
  int e = w / H, h = w - e * H;
  int s = src[e], d = dst[e];
  float eav = ea[e];
  const float* pl = xl + (size_t)s * 256 + h * C;
  const float* pr = xr + (size_t)d * 256 + h * C;
  const float* pe = We + h * C;
  const float* pa = att + h * C;
  float acc = 0.f;
  for (int c = lane; c < C; c += 32) {
    float m = pl[c] + pr[c] + eav * pe[c];
    m = (m > 0.f) ? m : 0.2f * m;                  // LeakyReLU(0.2)
    acc += m * pa[c];
  }
  acc = wave_sum(acc);
  if (lane == 0) {
    elog[w] = acc;
    atomicMax(emaxu + (size_t)d * H + h, ford(acc));
  }
}

__global__ void edge_expw_kernel(const int* __restrict__ dst,
                                 float* __restrict__ elog,
                                 const unsigned* __restrict__ emaxu,
                                 float* __restrict__ ewsum, int E, int H) {
  int t = blockIdx.x * blockDim.x + threadIdx.x;
  if (t >= E * H) return;
  int e = t / H, h = t - e * H;
  int d = dst[e];
  float w = __expf(elog[t] - funord(emaxu[(size_t)d * H + h]));
  elog[t] = w;
  atomicAdd(ewsum + (size_t)d * H + h, w);
}

__global__ void edge_scatter_kernel(const int* __restrict__ src, const int* __restrict__ dst,
                                    const float* __restrict__ elog, const float* __restrict__ ewsum,
                                    const float* __restrict__ xl, float* __restrict__ hacc,
                                    int E, int H, int C) {
  long long t = (long long)blockIdx.x * blockDim.x + threadIdx.x;
  long long total = (long long)E * 256;
  if (t >= total) return;
  int e = (int)(t >> 8);
  int hc = (int)(t & 255);
  int h = hc / C;
  int s = src[e], d = dst[e];
  float w = elog[(size_t)e * H + h];
  float alpha = w / (ewsum[(size_t)d * H + h] + 1e-16f);
  atomicAdd(hacc + (size_t)d * 256 + hc, alpha * xl[(size_t)s * 256 + hc]);
}

__global__ void finalize_kernel(float* __restrict__ hacc, const float* __restrict__ b,
                                unsigned short* __restrict__ hbf, long long total) {
  long long t = (long long)blockIdx.x * blockDim.x + threadIdx.x;
  if (t >= total) return;
  int hc = (int)(t & 255);
  float v = hacc[t] + b[hc];
  if (v < 0.f) v = 0.f;                            // ReLU after GAT layer
  hacc[t] = v;
  hbf[t] = f2bf(v);
}

// ---------- pooling ----------
// one wave32 per node: coalesced 256-channel dot + shuffle reduction
__global__ void pool_gate_kernel(const float* __restrict__ h, const float* __restrict__ gw,
                                 const float* __restrict__ gb, const int* __restrict__ batch,
                                 float* __restrict__ g, unsigned* __restrict__ gmaxu, int N) {
  int n = (int)((blockIdx.x * (long long)blockDim.x + threadIdx.x) >> 5);
  int lane = threadIdx.x & 31;
  if (n >= N) return;
  const float* p = h + (size_t)n * 256;
  float acc = 0.f;
  for (int c = lane; c < 256; c += 32) acc += p[c] * gw[c];
  acc = wave_sum(acc);
  if (lane == 0) {
    acc += gb[0];
    g[n] = acc;
    atomicMax(gmaxu + batch[n], ford(acc));
  }
}

__global__ void pool_w_kernel(const int* __restrict__ batch, float* __restrict__ g,
                              const unsigned* __restrict__ gmaxu, float* __restrict__ gws, int N) {
  int n = blockIdx.x * blockDim.x + threadIdx.x;
  if (n >= N) return;
  int b = batch[n];
  float w = __expf(g[n] - funord(gmaxu[b]));
  g[n] = w;
  atomicAdd(gws + b, w);
}

__global__ void pool_scatter_kernel(const int* __restrict__ batch, const float* __restrict__ g,
                                    const float* __restrict__ gws, const float* __restrict__ h,
                                    float* __restrict__ pooled, int N) {
  long long t = (long long)blockIdx.x * blockDim.x + threadIdx.x;
  long long total = (long long)N * 256;
  if (t >= total) return;
  int n = (int)(t >> 8), c = (int)(t & 255);
  int b = batch[n];
  atomicAdd(pooled + (size_t)b * 256 + c, (g[n] / (gws[b] + 1e-16f)) * h[t]);
}

// =====================================================================
extern "C" void kernel_launch(void* const* d_in, const int* in_sizes, int n_in,
                              void* d_out, int out_size, void* d_ws, size_t ws_size,
                              hipStream_t stream) {
  // top-level dict order; params flattened in sorted-key pytree order
  const float* x           = (const float*)d_in[0];
  const float* edge_attr   = (const float*)d_in[1];
  const float* agent_state = (const float*)d_in[2];
  const float* afc_b[4] = {(const float*)d_in[3], (const float*)d_in[5], (const float*)d_in[7], (const float*)d_in[9]};
  const float* afc_w[4] = {(const float*)d_in[4], (const float*)d_in[6], (const float*)d_in[8], (const float*)d_in[10]};
  const float* fo_b[3]  = {(const float*)d_in[11], (const float*)d_in[13], (const float*)d_in[15]};
  const float* fo_w[3]  = {(const float*)d_in[12], (const float*)d_in[14], (const float*)d_in[16]};
  const float* gWe[3]   = {(const float*)d_in[17], (const float*)d_in[22], (const float*)d_in[27]};
  const float* gWl[3]   = {(const float*)d_in[18], (const float*)d_in[23], (const float*)d_in[28]};
  const float* gWr[3]   = {(const float*)d_in[19], (const float*)d_in[24], (const float*)d_in[29]};
  const float* gatt[3]  = {(const float*)d_in[20], (const float*)d_in[25], (const float*)d_in[30]};
  const float* gbias[3] = {(const float*)d_in[21], (const float*)d_in[26], (const float*)d_in[31]};
  const float* gate_b   = (const float*)d_in[32];
  const float* gate_w   = (const float*)d_in[33];
  const float* gfc_b[4] = {(const float*)d_in[34], (const float*)d_in[36], (const float*)d_in[38], (const float*)d_in[40]};
  const float* gfc_w[4] = {(const float*)d_in[35], (const float*)d_in[37], (const float*)d_in[39], (const float*)d_in[41]};
  const int* src   = (const int*)d_in[42];
  const int* dstp  = src + NEDGES;
  const int* batch = (const int*)d_in[43];
  float* out = (float*)d_out;

  // ---- workspace carve-up ----
  char* ws = (char*)d_ws;
  size_t off = 0;
  auto take = [&](size_t bytes) -> void* {
    void* p = ws + off; off += (bytes + 255) & ~(size_t)255; return p;
  };
  unsigned short* hbf   = (unsigned short*)take((size_t)NNODES * 256 * 2);
  float* xl             = (float*)take((size_t)NNODES * 256 * 4);
  float* xr             = (float*)take((size_t)NNODES * 256 * 4);
  float* hacc           = (float*)take((size_t)NNODES * 256 * 4);
  unsigned short* wt    = (unsigned short*)take(256 * 256 * 2);
  float* elog           = (float*)take((size_t)NEDGES * 4 * 4);
  unsigned* emaxu       = (unsigned*)take((size_t)NNODES * 4 * 4);
  float* ewsum          = (float*)take((size_t)NNODES * 4 * 4);
  float* gbuf           = (float*)take((size_t)NNODES * 4);
  unsigned* gmaxu       = (unsigned*)take(NB * 4);
  float* gws            = (float*)take(NB * 4);
  float* pooled         = (float*)take((size_t)NB * 256 * 4);
  unsigned short* mbfA  = (unsigned short*)take((size_t)NB * 256 * 2);
  unsigned short* mbfB  = (unsigned short*)take((size_t)NB * 256 * 2);
  unsigned short* catbf = (unsigned short*)take((size_t)NB * 128 * 2);
  float* tf32           = (float*)take((size_t)NB * 256 * 4);

  const int gat_H[3]  = {4, 4, 2};
  const int gat_C[3]  = {64, 64, 128};
  const int gat_ic[3] = {4, 256, 256};
  const int gat_Kp[3] = {32, 256, 256};

  auto fill = [&](void* p, unsigned v, long long n) {
    fill_u32_kernel<<<4096, 256, 0, stream>>>((unsigned*)p, v, n);
  };

  // ---- layer-1 input: x [N x 4] -> bf16 [N x 32] ----
  cvt_pad_bf16_kernel<<<4096, 256, 0, stream>>>(x, hbf, NNODES, 4, 32);

  for (int L = 0; L < 3; ++L) {
    const int Kp = gat_Kp[L], H = gat_H[L], C = gat_C[L], ic = gat_ic[L];
    // xl = h @ Wl
    cvt_transpose_bf16_kernel<<<256, 256, 0, stream>>>(gWl[L], wt, ic, 256, 256, Kp);
    gemm_bf16_wmma_kernel<4><<<dim3(4, 80), 256, 0, stream>>>(
        hbf, wt, nullptr, xl, nullptr, NNODES, Kp, 256, 256, 0, 0);
    // xr = h @ Wr
    cvt_transpose_bf16_kernel<<<256, 256, 0, stream>>>(gWr[L], wt, ic, 256, 256, Kp);
    gemm_bf16_wmma_kernel<4><<<dim3(4, 80), 256, 0, stream>>>(
        hbf, wt, nullptr, xr, nullptr, NNODES, Kp, 256, 256, 0, 0);
    // segment softmax over edges
    fill(hacc, 0u, (long long)NNODES * 256);
    fill(emaxu, 0u, (long long)NNODES * H);   // 0 == ordered(-huge)
    fill(ewsum, 0u, (long long)NNODES * H);
    long long EHw = (long long)NEDGES * H * 32;   // one wave per (e,h)
    edge_logit_kernel<<<(unsigned)((EHw + 255) / 256), 256, 0, stream>>>(
        xl, xr, edge_attr, gWe[L], gatt[L], src, dstp, elog, emaxu, NEDGES, H, C);
    int EH = NEDGES * H;
    edge_expw_kernel<<<(EH + 255) / 256, 256, 0, stream>>>(dstp, elog, emaxu, ewsum, NEDGES, H);
    long long Etot = (long long)NEDGES * 256;
    edge_scatter_kernel<<<(unsigned)((Etot + 255) / 256), 256, 0, stream>>>(
        src, dstp, elog, ewsum, xl, hacc, NEDGES, H, C);
    // bias + relu, emit f32 (for pooling) and bf16 (next layer)
    long long Ntot = (long long)NNODES * 256;
    finalize_kernel<<<(unsigned)((Ntot + 255) / 256), 256, 0, stream>>>(hacc, gbias[L], hbf, Ntot);
  }

  // ---- global attention pooling ----
  fill(gmaxu, 0u, NB);
  fill(gws, 0u, NB);
  fill(pooled, 0u, (long long)NB * 256);
  long long Nw = (long long)NNODES * 32;          // one wave per node
  pool_gate_kernel<<<(unsigned)((Nw + 255) / 256), 256, 0, stream>>>(
      hacc, gate_w, gate_b, batch, gbuf, gmaxu, NNODES);
  pool_w_kernel<<<(NNODES + 255) / 256, 256, 0, stream>>>(batch, gbuf, gmaxu, gws, NNODES);
  long long Ptot = (long long)NNODES * 256;
  pool_scatter_kernel<<<(unsigned)((Ptot + 255) / 256), 256, 0, stream>>>(batch, gbuf, gws, hacc, pooled, NNODES);

  // ---- graph_fc MLP: 256->256->256->128->64 (relu on all but last) ----
  cvt_pad_bf16_kernel<<<64, 256, 0, stream>>>(pooled, mbfA, NB, 256, 256);
  cvt_transpose_bf16_kernel<<<256, 256, 0, stream>>>(gfc_w[0], wt, 256, 256, 256, 256);
  gemm_bf16_wmma_kernel<4><<<dim3(4, 1), 256, 0, stream>>>(mbfA, wt, gfc_b[0], tf32, mbfB, NB, 256, 256, 256, 256, 1);
  cvt_transpose_bf16_kernel<<<256, 256, 0, stream>>>(gfc_w[1], wt, 256, 256, 256, 256);
  gemm_bf16_wmma_kernel<4><<<dim3(4, 1), 256, 0, stream>>>(mbfB, wt, gfc_b[1], tf32, mbfA, NB, 256, 256, 256, 256, 1);
  cvt_transpose_bf16_kernel<<<128, 256, 0, stream>>>(gfc_w[2], wt, 256, 128, 128, 256);
  gemm_bf16_wmma_kernel<4><<<dim3(2, 1), 256, 0, stream>>>(mbfA, wt, gfc_b[2], tf32, mbfB, NB, 256, 128, 128, 128, 1);
  cvt_transpose_bf16_kernel<<<32, 256, 0, stream>>>(gfc_w[3], wt, 128, 64, 64, 128);
  gemm_bf16_wmma_kernel<4><<<dim3(1, 1), 256, 0, stream>>>(mbfB, wt, gfc_b[3], tf32, catbf, NB, 128, 64, 64, 128, 0);

  // ---- agent_fc MLP: 110->256->256->128->64 ----
  cvt_pad_bf16_kernel<<<32, 256, 0, stream>>>(agent_state, mbfA, NB, 110, 128);
  cvt_transpose_bf16_kernel<<<128, 256, 0, stream>>>(afc_w[0], wt, 110, 256, 256, 128);
  gemm_bf16_wmma_kernel<4><<<dim3(4, 1), 256, 0, stream>>>(mbfA, wt, afc_b[0], tf32, mbfB, NB, 128, 256, 256, 256, 1);
  cvt_transpose_bf16_kernel<<<256, 256, 0, stream>>>(afc_w[1], wt, 256, 256, 256, 256);
  gemm_bf16_wmma_kernel<4><<<dim3(4, 1), 256, 0, stream>>>(mbfB, wt, afc_b[1], tf32, mbfA, NB, 256, 256, 256, 256, 1);
  cvt_transpose_bf16_kernel<<<128, 256, 0, stream>>>(afc_w[2], wt, 256, 128, 128, 256);
  gemm_bf16_wmma_kernel<4><<<dim3(2, 1), 256, 0, stream>>>(mbfA, wt, afc_b[2], tf32, mbfB, NB, 256, 128, 128, 128, 1);
  cvt_transpose_bf16_kernel<<<32, 256, 0, stream>>>(afc_w[3], wt, 128, 64, 64, 128);
  gemm_bf16_wmma_kernel<4><<<dim3(1, 1), 256, 0, stream>>>(mbfB, wt, afc_b[3], tf32, catbf + 64, NB, 128, 64, 64, 128, 0);

  // ---- fc_out MLP on concat(gx, ax): 128->256->128->10 ----
  cvt_transpose_bf16_kernel<<<128, 256, 0, stream>>>(fo_w[0], wt, 128, 256, 256, 128);
  gemm_bf16_wmma_kernel<4><<<dim3(4, 1), 256, 0, stream>>>(catbf, wt, fo_b[0], tf32, mbfA, NB, 128, 256, 256, 256, 1);
  cvt_transpose_bf16_kernel<<<128, 256, 0, stream>>>(fo_w[1], wt, 256, 128, 128, 256);
  gemm_bf16_wmma_kernel<4><<<dim3(2, 1), 256, 0, stream>>>(mbfA, wt, fo_b[1], tf32, mbfB, NB, 256, 128, 128, 128, 1);
  cvt_transpose_bf16_kernel<<<16, 256, 0, stream>>>(fo_w[2], wt, 128, NA, 16, 128);
  gemm_bf16_wmma_kernel<1><<<dim3(1, 1), 256, 0, stream>>>(mbfB, wt, fo_b[2], out, nullptr, NB, 128, NA, NA, 0, 0);
}
